// LSTMCell_26714696581322
// MI455X (gfx1250) — matrix operand order repaired and checked by reading, not compiled, not run on hardware
//
#include <hip/hip_runtime.h>
#include <math.h>

// ---------------------------------------------------------------------------
// LSTM cell, fused noisy-weight + quantized-gate step for MI455X (gfx1250).
//   gates = [input|hx] (8192x1024, bf16) @ Wc (1024x2048, bf16)  via
//   v_wmma_f32_16x16x32_bf16, fp32 accumulate; gate math fused in epilogue.
// Main kernel: block = 8 waves; wave tile = 32 rows x 16 cols x 4 gates
//   -> 8 WMMAs per 12 b128 loads per K-step. Epilogue uses hardware
//   transcendentals (v_exp_f32) -- error << 1/256 quant grid.
// ---------------------------------------------------------------------------

typedef __attribute__((ext_vector_type(16))) __bf16 v16bf;
typedef __attribute__((ext_vector_type(8)))  __bf16 v8bf;
typedef __attribute__((ext_vector_type(8)))  float  v8f;

#define BATCH   8192
#define IN_SZ   512
#define H_SZ    512
#define K_SZ    1024          // IN + H (concatenated GEMM K)
#define G4      2048          // 4*H

// workspace layout (bytes)
#define WS_MAXES 0                                  // 4 floats
#define WS_BIAS  256                                // 2048 floats
#define WS_WC    8448                               // 2048*1024 bf16 = 4 MB
#define WS_AC    (8448 + (size_t)G4 * K_SZ * 2)     // 8192*1024 bf16 = 16 MB

// ---------------------------------------------------------------------------
// float atomic max (sign-split trick; dst pre-initialized to -inf bits)
// ---------------------------------------------------------------------------
__device__ __forceinline__ void atomicMaxF(float* a, float v) {
    if (v >= 0.0f) atomicMax((int*)a, __float_as_int(v));
    else           atomicMin((unsigned int*)a, __float_as_uint(v));
}

__global__ void init_max_kernel(float* maxes) {
    if (threadIdx.x < 4) maxes[threadIdx.x] = -INFINITY;
}

__global__ void reduce_max_kernel(const float* __restrict__ src, int n, float* dst) {
    float m = -INFINITY;
    for (int i = blockIdx.x * blockDim.x + threadIdx.x; i < n;
         i += gridDim.x * blockDim.x)
        m = fmaxf(m, src[i]);
    #pragma unroll
    for (int off = 16; off > 0; off >>= 1)
        m = fmaxf(m, __shfl_down(m, off, 32));
    __shared__ float smem[8];
    const int lane = threadIdx.x & 31, wave = threadIdx.x >> 5;
    if (lane == 0) smem[wave] = m;
    __syncthreads();
    if (wave == 0) {
        m = (lane < (int)(blockDim.x >> 5)) ? smem[lane] : -INFINITY;
        #pragma unroll
        for (int off = 4; off > 0; off >>= 1)
            m = fmaxf(m, __shfl_down(m, off, 32));
        if (lane == 0) atomicMaxF(dst, m);
    }
}

// ---------------------------------------------------------------------------
// Build bf16 noisy weight, K-contiguous per output column:
//   Wc[n][k] = w[n][k] + noise[k][n] * max(w)*0.1
// (B-fragment layout this code uses: lane = column, 16 consecutive K per
//  half-wave -> two contiguous b128 loads)
// ---------------------------------------------------------------------------
__global__ void pack_w_kernel(const float* __restrict__ wih, const float* __restrict__ whh,
                              const float* __restrict__ nih, const float* __restrict__ nhh,
                              const float* __restrict__ maxes, __bf16* __restrict__ Wc) {
    const int idx = blockIdx.x * blockDim.x + threadIdx.x;   // over 2048*1024
    const int n = idx >> 10;
    const int k = idx & 1023;
    float v;
    if (k < IN_SZ) v = wih[n * IN_SZ + k]           + nih[k * G4 + n]           * (maxes[0] * 0.1f);
    else           v = whh[n * H_SZ + (k - IN_SZ)]  + nhh[(k - IN_SZ) * G4 + n] * (maxes[1] * 0.1f);
    Wc[idx] = (__bf16)v;
}

__global__ void pack_bias_kernel(const float* __restrict__ bih, const float* __restrict__ bhh,
                                 const float* __restrict__ nbih, const float* __restrict__ nbhh,
                                 const float* __restrict__ maxes, float* __restrict__ biasc) {
    const int i = blockIdx.x * blockDim.x + threadIdx.x;     // over 2048
    biasc[i] = bih[i] + nbih[i] * (maxes[2] * 0.1f)
             + bhh[i] + nbhh[i] * (maxes[3] * 0.1f);
}

__global__ void pack_a_kernel(const float* __restrict__ input, const float* __restrict__ hx,
                              __bf16* __restrict__ Ac) {
    const int idx = blockIdx.x * blockDim.x + threadIdx.x;   // over 8192*1024
    const int b = idx >> 10;
    const int k = idx & 1023;
    const float v = (k < IN_SZ) ? input[b * IN_SZ + k] : hx[b * H_SZ + (k - IN_SZ)];
    Ac[idx] = (__bf16)v;
}

// ---------------------------------------------------------------------------
// fast device math (hardware v_exp_f32 path) + fake-quant helpers
// ---------------------------------------------------------------------------
__device__ __forceinline__ float fast_sigmoid(float x) {
    return 1.0f / (1.0f + __expf(-x));
}
__device__ __forceinline__ float fast_tanh(float x) {
    // 1 - 2/(e^{2x}+1): saturates correctly (+inf -> 1, -inf -> -1)
    return 1.0f - 2.0f / (__expf(2.0f * x) + 1.0f);
}
__device__ __forceinline__ float fq_u8(float x) {        // unsigned, 8 bits
    float xc = fminf(fmaxf(x, 0.0f), 1.0f);
    return rintf(xc * 256.0f) * (1.0f / 256.0f);
}
__device__ __forceinline__ float fq_s8(float x) {        // signed, 8 bits
    const float d = 0.0078125f;                          // 1/128
    float xc = fminf(fmaxf(x, -1.0f + d), 1.0f - d);
    return rintf(xc * 128.0f) * d;
}

__device__ __forceinline__ v8f wmma_bf16(v16bf a, v16bf b, v8f c) {
    return __builtin_amdgcn_wmma_f32_16x16x32_bf16(false, a, false, b,
                                                   (short)0, c, false, false);
}

// ---------------------------------------------------------------------------
// Main fused GEMM + gate kernel.
// Grid: (BATCH/256, H/16). Block: 256 threads = 8 waves.
// Wave w: rows [blockIdx.x*256 + w*32, +32) as two 16-row strips,
// columns h in [blockIdx.y*16, +16), 4 gate column-blocks.
// ---------------------------------------------------------------------------
__global__ __launch_bounds__(256)
void lstm_wmma_kernel(const __bf16* __restrict__ Ac, const __bf16* __restrict__ Wc,
                      const float* __restrict__ biasc, const float* __restrict__ cx,
                      float* __restrict__ hy, float* __restrict__ cy_out) {
    const int lane  = threadIdx.x & 31;
    const int wave  = threadIdx.x >> 5;
    const int half  = lane >> 4;        // half-wave select
    const int nloc  = lane & 15;        // column within the 16-wide tile
    const int mbase = blockIdx.x * 256 + wave * 32;
    const int n0    = blockIdx.y * 16;

    // acc[g][s]: gate g, row-strip s (rows mbase + s*16 ..)
    v8f acc00 = {}, acc01 = {};   // gate 0 (ingate)
    v8f acc10 = {}, acc11 = {};   // gate 1 (forget)
    v8f acc20 = {}, acc21 = {};   // gate 2 (cell)
    v8f acc30 = {}, acc31 = {};   // gate 3 (out)

    // A fragment bases: row = mbase (+16) + nloc; half-wave K split 0-15 vs 16-31
    const __bf16* a0p = Ac + (size_t)(mbase + nloc) * K_SZ + half * 16;
    const __bf16* a1p = a0p + (size_t)16 * K_SZ;
    // B fragment bases: lane = weight row (output column), 16 consecutive K per half
    const __bf16* b0p = Wc + (size_t)(0 * H_SZ + n0 + nloc) * K_SZ + half * 16;
    const __bf16* b1p = Wc + (size_t)(1 * H_SZ + n0 + nloc) * K_SZ + half * 16;
    const __bf16* b2p = Wc + (size_t)(2 * H_SZ + n0 + nloc) * K_SZ + half * 16;
    const __bf16* b3p = Wc + (size_t)(3 * H_SZ + n0 + nloc) * K_SZ + half * 16;

    #pragma unroll 2
    for (int k0 = 0; k0 < K_SZ; k0 += 32) {
        union { v16bf v; v8bf h[2]; } a0, a1;
        a0.h[0] = *(const v8bf*)(a0p + k0);            // K = k0 + half*16 + [0..7]
        a0.h[1] = *(const v8bf*)(a0p + k0 + 8);        // K = k0 + half*16 + [8..15]
        a1.h[0] = *(const v8bf*)(a1p + k0);
        a1.h[1] = *(const v8bf*)(a1p + k0 + 8);
        const v16bf b0 = *(const v16bf*)(b0p + k0);    // K = k0 + half*16 + [0..15]
        const v16bf b1 = *(const v16bf*)(b1p + k0);
        const v16bf b2 = *(const v16bf*)(b2p + k0);
        const v16bf b3 = *(const v16bf*)(b3p + k0);
        acc00 = wmma_bf16(a0.v, b0, acc00);
        acc01 = wmma_bf16(a1.v, b0, acc01);
        acc10 = wmma_bf16(a0.v, b1, acc10);
        acc11 = wmma_bf16(a1.v, b1, acc11);
        acc20 = wmma_bf16(a0.v, b2, acc20);
        acc21 = wmma_bf16(a1.v, b2, acc21);
        acc30 = wmma_bf16(a0.v, b3, acc30);
        acc31 = wmma_bf16(a1.v, b3, acc31);
    }

    // Epilogue: C/D layout -> lane VGPR r holds (row = strip + half*8 + r, col = n0+nloc)
    const int h = n0 + nloc;
    const float bi = biasc[0 * H_SZ + h];
    const float bf = biasc[1 * H_SZ + h];
    const float bc = biasc[2 * H_SZ + h];
    const float bo = biasc[3 * H_SZ + h];

    #pragma unroll
    for (int s = 0; s < 2; ++s) {
        const v8f& g0 = s ? acc01 : acc00;
        const v8f& g1 = s ? acc11 : acc10;
        const v8f& g2 = s ? acc21 : acc20;
        const v8f& g3 = s ? acc31 : acc30;
        #pragma unroll
        for (int r = 0; r < 8; ++r) {
            const int row = mbase + s * 16 + half * 8 + r;
            const float ig = fq_u8(fast_sigmoid(g0[r] + bi));
            const float fg = fq_u8(fast_sigmoid(g1[r] + bf));
            const float cg = fq_s8(fast_tanh   (g2[r] + bc));
            const float og = fq_u8(fast_sigmoid(g3[r] + bo));
            const float cp = cx[(size_t)row * H_SZ + h];
            const float cy = fq_s8((fq_s8(fg * cp) + fq_s8(ig * cg)) * 0.5f);
            const float hv = fq_s8(og * fq_s8(fast_tanh(cy * 2.0f)));
            hy    [(size_t)row * H_SZ + h] = hv;
            cy_out[(size_t)row * H_SZ + h] = cy;
        }
    }
}

// ---------------------------------------------------------------------------
extern "C" void kernel_launch(void* const* d_in, const int* in_sizes, int n_in,
                              void* d_out, int out_size, void* d_ws, size_t ws_size,
                              hipStream_t stream) {
    const float* input = (const float*)d_in[0];
    const float* hx    = (const float*)d_in[1];
    const float* cx    = (const float*)d_in[2];
    const float* w_ih  = (const float*)d_in[3];
    const float* w_hh  = (const float*)d_in[4];
    const float* b_ih  = (const float*)d_in[5];
    const float* b_hh  = (const float*)d_in[6];
    const float* n_ih  = (const float*)d_in[7];
    const float* n_hh  = (const float*)d_in[8];
    const float* nb_ih = (const float*)d_in[9];
    const float* nb_hh = (const float*)d_in[10];

    char*   ws    = (char*)d_ws;
    float*  maxes = (float*)(ws + WS_MAXES);
    float*  biasc = (float*)(ws + WS_BIAS);
    __bf16* Wc    = (__bf16*)(ws + WS_WC);
    __bf16* Ac    = (__bf16*)(ws + WS_AC);

    float* hy = (float*)d_out;
    float* cy = hy + (size_t)BATCH * H_SZ;

    init_max_kernel<<<1, 32, 0, stream>>>(maxes);
    reduce_max_kernel<<<256, 256, 0, stream>>>(w_ih, G4 * IN_SZ, maxes + 0);
    reduce_max_kernel<<<256, 256, 0, stream>>>(w_hh, G4 * H_SZ,  maxes + 1);
    reduce_max_kernel<<<8,   256, 0, stream>>>(b_ih, G4,         maxes + 2);
    reduce_max_kernel<<<8,   256, 0, stream>>>(b_hh, G4,         maxes + 3);

    pack_w_kernel   <<<(G4 * K_SZ) / 256,    256, 0, stream>>>(w_ih, w_hh, n_ih, n_hh, maxes, Wc);
    pack_bias_kernel<<<G4 / 256,             256, 0, stream>>>(b_ih, b_hh, nb_ih, nb_hh, maxes, biasc);
    pack_a_kernel   <<<(BATCH * K_SZ) / 256, 256, 0, stream>>>(input, hx, Ac);

    dim3 grid(BATCH / 256, H_SZ / 16);
    lstm_wmma_kernel<<<grid, 256, 0, stream>>>(Ac, Wc, biasc, cx, hy, cy);
}
